// GCN_62818191671411
// MI455X (gfx1250) — compile-verified, matrix operand order checked by reading, not run on hardware
//
#include <hip/hip_runtime.h>
#include <hip/hip_bf16.h>
#include <stdint.h>

typedef _Float16 f16;
typedef __attribute__((ext_vector_type(16))) _Float16 v16h;
typedef __attribute__((ext_vector_type(8)))  _Float16 f16x8;
typedef __attribute__((ext_vector_type(4)))  _Float16 f16x4;
typedef __attribute__((ext_vector_type(8)))  float    v8f;

#define IN_DIM  512
#define HID_DIM 256
#define NCLS    64

// Load a 16-bit WMMA A/B fragment for one lane.
// p points at element (k0 + half*8) of a K-contiguous row/col; second chunk at +16.
__device__ __forceinline__ v16h load_frag16(const f16* p) {
  union { v16h v; f16x8 h[2]; } u;
  u.h[0] = *(const f16x8*)(p);
  u.h[1] = *(const f16x8*)(p + 16);
  return u.v;
}

// ---------------- degree / normalization ----------------
__global__ void k_init_deg(float* deg, int n) {
  int i = blockIdx.x * 256 + threadIdx.x;
  if (i < n) deg[i] = 1.0f;                 // self-loop
}

__global__ void k_count_deg(const int* __restrict__ col, float* deg, int E) {
  int e = blockIdx.x * 256 + threadIdx.x;
  if (e < E) atomicAdd(&deg[col[e]], 1.0f);
}

__global__ void k_dinv(float* deg, int n) {
  int i = blockIdx.x * 256 + threadIdx.x;
  if (i < n) {
    float d = deg[i];
    deg[i] = (d > 0.0f) ? rsqrtf(d) : 0.0f; // in-place: deg becomes dinv
  }
}

// ---------------- precision conversion / transpose ----------------
__global__ void k_cvt_x(const float* __restrict__ x, f16* __restrict__ xh, int total4) {
  int i = blockIdx.x * 256 + threadIdx.x;
  if (i < total4) {
    float4 v = ((const float4*)x)[i];
    f16x4 o;
    o[0] = (f16)v.x; o[1] = (f16)v.y; o[2] = (f16)v.z; o[3] = (f16)v.w;
    ((f16x4*)xh)[i] = o;
  }
}

// W [K][Ncol] (row-major) -> Wt [Ncol][K] fp16 (K contiguous per output col)
__global__ void k_trans_w(const float* __restrict__ W, f16* __restrict__ Wt, int K, int Ncol) {
  int idx = blockIdx.x * 256 + threadIdx.x;
  if (idx < K * Ncol) {
    int n = idx / K;
    int k = idx - n * K;
    Wt[idx] = (f16)W[(size_t)k * Ncol + n];
  }
}

// ---------------- GEMM1: xw = x @ W1 ; agg1 = xw * dinv^2 (self-loop seed) ----------------
__global__ __launch_bounds__(128) void k_gemm1(const f16* __restrict__ xh,
                                               const f16* __restrict__ w1t,
                                               const float* __restrict__ dinv,
                                               float* __restrict__ xw,
                                               float* __restrict__ agg1, int n) {
  const int wave = threadIdx.x >> 5;
  const int lane = threadIdx.x & 31;
  const int half = lane >> 4;     // 0: lanes 0-15, 1: lanes 16-31
  const int lp   = lane & 15;
  const int rowBase = blockIdx.x * 64 + wave * 16;
  const int colBase = blockIdx.y * 64;

  int m = rowBase + lp;
  if (m >= n) m = n - 1;          // clamp loads; stores are guarded

  v8f acc[4] = {v8f{}, v8f{}, v8f{}, v8f{}};
  const f16* aptr = xh + (size_t)m * IN_DIM + half * 8;
  const f16* bptr[4];
#pragma unroll
  for (int nt = 0; nt < 4; ++nt)
    bptr[nt] = w1t + (size_t)(colBase + nt * 16 + lp) * IN_DIM + half * 8;

#pragma unroll 2
  for (int k0 = 0; k0 < IN_DIM; k0 += 32) {
    v16h a = load_frag16(aptr + k0);
#pragma unroll
    for (int nt = 0; nt < 4; ++nt) {
      v16h b = load_frag16(bptr[nt] + k0);
      acc[nt] = __builtin_amdgcn_wmma_f32_16x16x32_f16(false, a, false, b,
                                                       (short)0, acc[nt], false, false);
    }
  }

#pragma unroll
  for (int v = 0; v < 8; ++v) {
    int r = rowBase + half * 8 + v;       // C/D layout: VGPR v -> M = v + 8*half
    if (r < n) {
      float di = dinv[r];
      float s  = di * di;
      size_t o = (size_t)r * HID_DIM;
#pragma unroll
      for (int nt = 0; nt < 4; ++nt) {
        int c = colBase + nt * 16 + lp;   // C/D layout: N = lane&15
        float val = acc[nt][v];
        xw[o + c]   = val;
        agg1[o + c] = val * s;
      }
    }
  }
}

// ---------------- edge scatter layer 1: agg1[c] += xw[r] * dinv[r]*dinv[c] ----------------
__global__ __launch_bounds__(256) void k_scatter1(const int* __restrict__ row,
                                                  const int* __restrict__ col,
                                                  const float* __restrict__ dinv,
                                                  const float* __restrict__ xw,
                                                  float* __restrict__ agg, int E) {
  int t = blockIdx.x * 256 + threadIdx.x;
  int e = t >> 5;                 // one edge per 32 lanes, 8 floats per lane
  if (e >= E) return;
  int lane = t & 31;
  int r = row[e];
  int c = col[e];
  const float4* src = (const float4*)(xw + (size_t)r * HID_DIM) + lane * 2;
  __builtin_prefetch(src, 0, 3);  // global_prefetch: pull gather row toward L0
  float nrm = dinv[r] * dinv[c];
  float* dst = agg + (size_t)c * HID_DIM + lane * 8;
  float4 v0 = src[0];
  float4 v1 = src[1];
  atomicAdd(dst + 0, v0.x * nrm); atomicAdd(dst + 1, v0.y * nrm);
  atomicAdd(dst + 2, v0.z * nrm); atomicAdd(dst + 3, v0.w * nrm);
  atomicAdd(dst + 4, v1.x * nrm); atomicAdd(dst + 5, v1.y * nrm);
  atomicAdd(dst + 6, v1.z * nrm); atomicAdd(dst + 7, v1.w * nrm);
}

// ---------------- GEMM2: hw = f(agg1) @ W2 ; out = hw * dinv^2 + b2 ----------------
// f = relu(agg1 + b1) * mask * 2   (dropout p=0.5), fused into the A-fragment load
// with fully vectorized (b128) loads of agg1 / mask / b1.
__global__ __launch_bounds__(128) void k_gemm2(const float* __restrict__ agg1,
                                               const int* __restrict__ mask,
                                               const float* __restrict__ b1,
                                               const f16* __restrict__ w2t,
                                               const float* __restrict__ b2,
                                               const float* __restrict__ dinv,
                                               float* __restrict__ hw,
                                               float* __restrict__ outp, int n) {
  const int wave = threadIdx.x >> 5;
  const int lane = threadIdx.x & 31;
  const int half = lane >> 4;
  const int lp   = lane & 15;
  const int rowBase = blockIdx.x * 64 + wave * 16;

  int m = rowBase + lp;
  if (m >= n) m = n - 1;

  v8f acc[4] = {v8f{}, v8f{}, v8f{}, v8f{}};
  const float* ap = agg1 + (size_t)m * HID_DIM;
  const int*   mp = mask + (size_t)m * HID_DIM;

#pragma unroll 2
  for (int k0 = 0; k0 < HID_DIM; k0 += 32) {
    const int kb = k0 + half * 8;
    union { v16h v; f16 e[16]; } ua;
#pragma unroll
    for (int c = 0; c < 2; ++c) {          // chunk 0: [kb..kb+7], chunk 1: [kb+16..kb+23]
      const int kk = kb + c * 16;
      float4 a0 = *(const float4*)(ap + kk);
      float4 a1 = *(const float4*)(ap + kk + 4);
      int4   m0 = *(const int4*)(mp + kk);
      int4   m1 = *(const int4*)(mp + kk + 4);
      float4 g0 = *(const float4*)(b1 + kk);
      float4 g1 = *(const float4*)(b1 + kk + 4);
      float t0 = fmaxf(a0.x + g0.x, 0.0f) * (m0.x ? 2.0f : 0.0f);
      float t1 = fmaxf(a0.y + g0.y, 0.0f) * (m0.y ? 2.0f : 0.0f);
      float t2 = fmaxf(a0.z + g0.z, 0.0f) * (m0.z ? 2.0f : 0.0f);
      float t3 = fmaxf(a0.w + g0.w, 0.0f) * (m0.w ? 2.0f : 0.0f);
      float t4 = fmaxf(a1.x + g1.x, 0.0f) * (m1.x ? 2.0f : 0.0f);
      float t5 = fmaxf(a1.y + g1.y, 0.0f) * (m1.y ? 2.0f : 0.0f);
      float t6 = fmaxf(a1.z + g1.z, 0.0f) * (m1.z ? 2.0f : 0.0f);
      float t7 = fmaxf(a1.w + g1.w, 0.0f) * (m1.w ? 2.0f : 0.0f);
      ua.e[c * 8 + 0] = (f16)t0; ua.e[c * 8 + 1] = (f16)t1;
      ua.e[c * 8 + 2] = (f16)t2; ua.e[c * 8 + 3] = (f16)t3;
      ua.e[c * 8 + 4] = (f16)t4; ua.e[c * 8 + 5] = (f16)t5;
      ua.e[c * 8 + 6] = (f16)t6; ua.e[c * 8 + 7] = (f16)t7;
    }
#pragma unroll
    for (int nt = 0; nt < 4; ++nt) {
      v16h b = load_frag16(w2t + (size_t)(nt * 16 + lp) * HID_DIM + kb);
      acc[nt] = __builtin_amdgcn_wmma_f32_16x16x32_f16(false, ua.v, false, b,
                                                       (short)0, acc[nt], false, false);
    }
  }

#pragma unroll
  for (int v = 0; v < 8; ++v) {
    int r = rowBase + half * 8 + v;
    if (r < n) {
      float di = dinv[r];
      float s  = di * di;
      size_t o = (size_t)r * NCLS;
#pragma unroll
      for (int nt = 0; nt < 4; ++nt) {
        int c = nt * 16 + lp;
        float val = acc[nt][v];
        hw[o + c]   = val;
        outp[o + c] = val * s + b2[c];           // self-loop + bias seed
      }
    }
  }
}

// ---------------- edge scatter layer 2: out[c] += hw[r] * dinv[r]*dinv[c] ----------------
__global__ __launch_bounds__(256) void k_scatter2(const int* __restrict__ row,
                                                  const int* __restrict__ col,
                                                  const float* __restrict__ dinv,
                                                  const float* __restrict__ hw,
                                                  float* __restrict__ outp, int E) {
  int t = blockIdx.x * 256 + threadIdx.x;
  int e = t >> 4;                 // one edge per 16 lanes, 4 floats per lane
  if (e >= E) return;
  int lane = t & 15;
  int r = row[e];
  int c = col[e];
  const float4* src = (const float4*)(hw + (size_t)r * NCLS) + lane;
  __builtin_prefetch(src, 0, 3);
  float nrm = dinv[r] * dinv[c];
  float4 v = *src;
  float* dst = outp + (size_t)c * NCLS + lane * 4;
  atomicAdd(dst + 0, v.x * nrm); atomicAdd(dst + 1, v.y * nrm);
  atomicAdd(dst + 2, v.z * nrm); atomicAdd(dst + 3, v.w * nrm);
}

extern "C" void kernel_launch(void* const* d_in, const int* in_sizes, int n_in,
                              void* d_out, int out_size, void* d_ws, size_t ws_size,
                              hipStream_t stream) {
  const float* x    = (const float*)d_in[0];
  const int*   ei   = (const int*)d_in[1];
  const int*   mask = (const int*)d_in[2];
  const float* W1   = (const float*)d_in[3];
  const float* b1   = (const float*)d_in[4];
  const float* W2   = (const float*)d_in[5];
  const float* b2   = (const float*)d_in[6];
  float* outp = (float*)d_out;

  const int N = in_sizes[0] / IN_DIM;   // 50000
  const int E = in_sizes[1] / 2;        // 800000
  const int* row = ei;                  // edge_index[0] = source
  const int* col = ei + E;              // edge_index[1] = target

  // workspace carve-out (256B aligned regions)
  char* wsb = (char*)d_ws;
  size_t off = 0;
  auto carve = [&](size_t bytes) -> char* {
    char* p = wsb + off;
    off = (off + bytes + 255) & ~(size_t)255;
    return p;
  };
  float* deg  = (float*)carve((size_t)N * 4);                  // becomes dinv
  f16*   xh   = (f16*)  carve((size_t)N * IN_DIM * 2);
  f16*   w1t  = (f16*)  carve((size_t)IN_DIM * HID_DIM * 2);
  f16*   w2t  = (f16*)  carve((size_t)HID_DIM * NCLS * 2);
  float* xw   = (float*)carve((size_t)N * HID_DIM * 4);
  float* agg1 = (float*)carve((size_t)N * HID_DIM * 4);
  float* hw   = (float*)carve((size_t)N * NCLS * 4);
  (void)ws_size; (void)n_in; (void)out_size;

  // 1) degrees + normalization
  k_init_deg<<<(N + 255) / 256, 256, 0, stream>>>(deg, N);
  k_count_deg<<<(E + 255) / 256, 256, 0, stream>>>(col, deg, E);
  k_dinv<<<(N + 255) / 256, 256, 0, stream>>>(deg, N);

  // 2) fp16 staging
  int x4 = (N * IN_DIM) / 4;
  k_cvt_x<<<(x4 + 255) / 256, 256, 0, stream>>>(x, xh, x4);
  k_trans_w<<<(IN_DIM * HID_DIM + 255) / 256, 256, 0, stream>>>(W1, w1t, IN_DIM, HID_DIM);
  k_trans_w<<<(HID_DIM * NCLS + 255) / 256, 256, 0, stream>>>(W2, w2t, HID_DIM, NCLS);

  // 3) layer 1: GEMM (seeds self-loop) + edge scatter
  dim3 g1((N + 63) / 64, HID_DIM / 64);
  k_gemm1<<<g1, 128, 0, stream>>>(xh, w1t, deg, xw, agg1, N);
  k_scatter1<<<(E * 32 + 255) / 256, 256, 0, stream>>>(row, col, deg, xw, agg1, E);

  // 4) layer 2: fused relu/dropout GEMM (seeds self-loop + b2) + edge scatter
  k_gemm2<<<(N + 63) / 64, 128, 0, stream>>>(agg1, mask, b1, w2t, b2, deg, hw, outp, N);
  k_scatter2<<<(E * 16 + 255) / 256, 256, 0, stream>>>(row, col, deg, hw, outp, E);
}
